// NeuralJumpSDE_75737453298058
// MI455X (gfx1250) — matrix operand order repaired and checked by reading, not compiled, not run on hardware
//
#include <hip/hip_runtime.h>
#include <hip/hip_bf16.h>

typedef __attribute__((ext_vector_type(16))) _Float16 v16h;
typedef __attribute__((ext_vector_type(8)))  float    v8f;

#define LOG_2PI 1.8378770664093453f

struct Params {
  const _Float16 *WF1,*WF2,*WF3,*WG1,*WG2,*WG3,*WW1,*WW2,*WW3,*WL1,*WL2,*WL3;
  const float *bF1,*bF2,*bF3,*bG1,*bG2,*bG3,*bW1,*bW2,*bW3,*bL1,*bL2,*bL3;
  const float *obs, *c0, *dtp;
  const int   *nsp;
  int T;
};

// ---------------------------------------------------------------------------
// Weight conversion: fp32 row-major (out,in) -> f16 WMMA A-fragment tiles.
// Tile (blk,c) covers rows [blk*16,blk*16+16), K-cols [c*32,c*32+32).
// Within a tile: lane L (m=L&15, h=L>>4), half j:
//   K = h*16 + j      (contiguous order, matches B fragment)
// Stored contiguously: dst[tile*512 + L*16 + j]  -> 32B/lane coalesced loads.
// ---------------------------------------------------------------------------
__global__ void convert_weights(const float* __restrict__ src, _Float16* __restrict__ dst,
                                int outDim, int inDim, int outP, int inP) {
  int gid = blockIdx.x * blockDim.x + threadIdx.x;
  int total = outP * inP;
  if (gid >= total) return;
  int tile   = gid >> 9;
  int within = gid & 511;
  int lane   = within >> 4;
  int j      = within & 15;
  int nk     = inP >> 5;
  int blk    = tile / nk;
  int c      = tile - blk * nk;
  int h      = lane >> 4;
  int row    = (blk << 4) + (lane & 15);
  int col    = (c << 5) + h * 16 + j;
  float v = (row < outDim && col < inDim) ? src[row * inDim + col] : 0.f;
  dst[gid] = (_Float16)v;
}

// ---------------------------------------------------------------------------
// Wave32 reduction helpers
// ---------------------------------------------------------------------------
__device__ __forceinline__ float wave_sum(float v) {
  #pragma unroll
  for (int m = 16; m >= 1; m >>= 1) v += __shfl_xor(v, m, 32);
  return v;
}
__device__ __forceinline__ float half16_sum(float v) { // lanes 16-31 mirror 0-15
  #pragma unroll
  for (int m = 8; m >= 1; m >>= 1) v += __shfl_xor(v, m, 32);
  return v;
}
__device__ __forceinline__ float half16_max(float v) {
  #pragma unroll
  for (int m = 8; m >= 1; m >>= 1) v = fmaxf(v, __shfl_xor(v, m, 32));
  return v;
}
__device__ __forceinline__ float softplusf(float x) {
  return fmaxf(x, 0.f) + log1pf(__expf(-fabsf(x)));
}

// ---------------------------------------------------------------------------
// WMMA matvec: act(W @ x + b) with x already f16 in LDS (B-fragment layout:
// lane-half h holds K = h*16 + j, identical across N since all 16 columns
// broadcast the same x). A-fragment: one 32B/lane contiguous global load.
// Output goes to f32 LDS (y32) and/or f16 LDS (y16) from lanes (L&15)==0,
// which own C rows r + 8*(L>=16) in acc[r]. Entry __syncthreads() publishes
// the producer's x; WAR on reused buffers spans >= 2 syncs by schedule.
// Runs with 16 waves (blockDim 512); WMMA executed under full EXEC.
// ---------------------------------------------------------------------------
template<int INP, int OUTP>
__device__ __forceinline__ void matvec(const _Float16* __restrict__ Wt,
                                       const _Float16* __restrict__ x,
                                       float* __restrict__ y32,
                                       _Float16* __restrict__ y16,
                                       const float* __restrict__ bias,
                                       int outValid, bool act) {
  __syncthreads();
  const int tid  = threadIdx.x;
  const int lane = tid & 31;
  const int wave = tid >> 5;
  const int h    = lane >> 4;
  constexpr int NBLK = OUTP >> 4;
  constexpr int NK   = INP  >> 5;
  for (int blk = wave; blk < NBLK; blk += 16) {
    v8f acc = {};
    const _Float16* wt = Wt + ((size_t)(blk * NK) << 9) + (lane << 4);
    #pragma unroll
    for (int c = 0; c < NK; ++c) {
      v16h a = *(const v16h*)wt;                       // 2x global_load_b128
      __builtin_prefetch(wt + 512, 0, 1);
      wt += 512;
      v16h b = *(const v16h*)(x + (c << 5) + (h << 4)); // 2x ds_load_b128
      acc = __builtin_amdgcn_wmma_f32_16x16x32_f16(false, a, false, b,
                                                   (short)0, acc, false, false);
    }
    if ((lane & 15) == 0) {
      const int rbase = (blk << 4) + (h << 3);
      #pragma unroll
      for (int r = 0; r < 8; ++r) {
        int row = rbase + r;
        if (row < outValid) {
          float v = acc[r] + bias[row];
          if (act) v = (v > 0.f) ? v : (__expf(v) - 1.f);  // celu(alpha=1)
          if (y32) y32[row] = v;
          if (y16) y16[row] = (_Float16)v;
        }
      }
    }
  }
}

// ---------------------------------------------------------------------------
// ODE right-hand side: kout = f(zin)   (zin fp32, staged to f16 once)
// ---------------------------------------------------------------------------
__device__ __forceinline__ void ode_fn(const Params& p,
    const float* zin, float* kout,
    _Float16* x16, _Float16* actA16, _Float16* actB16,
    float* gtmp, float* red) {
  const int tid = threadIdx.x;
  if (tid < 256) x16[tid] = (_Float16)zin[tid];     // published by matvec sync
  // dcdt = F(z)
  matvec<256,256>(p.WF1, x16,    nullptr, actA16, p.bF1, 256, true);
  matvec<256,256>(p.WF2, actA16, nullptr, actB16, p.bF2, 256, true);
  matvec<256,128>(p.WF3, actB16, kout,   nullptr, p.bF3, 128, false);
  // g = G(c)  (reads x16[0:128] = c)
  matvec<128,256>(p.WG1, x16,    nullptr, actA16, p.bG1, 256, true);
  matvec<256,256>(p.WG2, actA16, nullptr, actB16, p.bG2, 256, true);
  matvec<256,128>(p.WG3, actB16, gtmp,   nullptr, p.bG3, 128, false);
  __syncthreads();
  if (tid < 32) {                       // wave 0: c.c and dcdt.c over 128 elems
    float cs = 0.f, ds = 0.f;
    #pragma unroll
    for (int i = 0; i < 4; ++i) {
      float cv = zin[tid + 32 * i];
      cs += cv * cv;
      ds += kout[tid + 32 * i] * cv;
    }
    cs = wave_sum(cs);
    ds = wave_sum(ds);
    if (tid == 0) red[0] = ds / (cs + 1e-8f);
  }
  __syncthreads();
  const float s = red[0];
  if (tid < 128) {
    kout[tid] -= s * zin[tid];                              // tangent projection
  } else if (tid < 256) {
    const int i = tid - 128;
    kout[128 + i] = -softplusf(gtmp[i]) * zin[128 + i];     // dh/dt
  }
  __syncthreads();
}

// ---------------------------------------------------------------------------
// z[128:256] += W_mlp(concat(c, y))
// ---------------------------------------------------------------------------
__device__ __forceinline__ void apply_jump(const Params& p, const float* __restrict__ y,
    float* z, _Float16* x16, _Float16* actA16, _Float16* actB16, float* dh) {
  const int tid = threadIdx.x;
  if (tid < 160) {                                  // zero-padded 132 -> 160
    float v = 0.f;
    if (tid < 128)       v = z[tid];
    else if (tid < 132)  v = y[tid - 128];
    x16[tid] = (_Float16)v;
  }
  matvec<160,256>(p.WW1, x16,    nullptr, actA16, p.bW1, 256, true);
  matvec<256,256>(p.WW2, actA16, nullptr, actB16, p.bW2, 256, true);
  matvec<256,128>(p.WW3, actB16, dh,     nullptr, p.bW3, 128, false);
  __syncthreads();
  if (tid < 128) z[128 + tid] += dh[tid];
  __syncthreads();
}

// ---------------------------------------------------------------------------
// Persistent single-workgroup scan kernel (1 WGP, 16 wave32s).
// ---------------------------------------------------------------------------
__global__ __launch_bounds__(512)
void njsde_main(Params p, float* __restrict__ out) {
  __shared__ float z[256], ztmp[256], kbuf[256], ksum[256];
  __shared__ float gtmp[128], dh[128];
  __shared__ float lout[160];
  __shared__ float red[1];
  __shared__ alignas(32) _Float16 x16[256];
  __shared__ alignas(32) _Float16 actA16[256];
  __shared__ alignas(32) _Float16 actB16[256];
  const int tid = threadIdx.x;

  if (tid < 256) z[tid] = (tid < 128) ? p.c0[tid] : 0.f;
  __syncthreads();

  const float dt  = *p.dtp;
  const int   ns  = *p.nsp;
  const float sub = dt / (float)ns;

  float logp_sum = 0.f;

  apply_jump(p, p.obs, z, x16, actA16, actB16, dh);   // initial jump with obs[0]

  for (int t = 1; t < p.T; ++t) {
    // ---- integrate: ns RK4 sub-steps ----
    for (int s = 0; s < ns; ++s) {
      ode_fn(p, z, kbuf, x16, actA16, actB16, gtmp, red);              // k1
      if (tid < 256) { ksum[tid]  = kbuf[tid];
                       ztmp[tid]  = fmaf(0.5f * sub, kbuf[tid], z[tid]); }
      __syncthreads();
      ode_fn(p, ztmp, kbuf, x16, actA16, actB16, gtmp, red);           // k2
      if (tid < 256) { ksum[tid] += 2.f * kbuf[tid];
                       ztmp[tid]  = fmaf(0.5f * sub, kbuf[tid], z[tid]); }
      __syncthreads();
      ode_fn(p, ztmp, kbuf, x16, actA16, actB16, gtmp, red);           // k3
      if (tid < 256) { ksum[tid] += 2.f * kbuf[tid];
                       ztmp[tid]  = fmaf(sub, kbuf[tid], z[tid]); }
      __syncthreads();
      ode_fn(p, ztmp, kbuf, x16, actA16, actB16, gtmp, red);           // k4
      if (tid < 256) z[tid] += (sub * (1.f / 6.f)) * (ksum[tid] + kbuf[tid]);
      __syncthreads();
    }
    // ---- GMM log-likelihood: lout = L(z) (144 valid of 160) ----
    const float* y = p.obs + (size_t)t * 4;
    if (tid < 256) x16[tid] = (_Float16)z[tid];
    matvec<256,256>(p.WL1, x16,    nullptr, actA16, p.bL1, 256, true);
    matvec<256,256>(p.WL2, actA16, nullptr, actB16, p.bL2, 256, true);
    matvec<256,160>(p.WL3, actB16, lout,   nullptr, p.bL3, 144, false);
    __syncthreads();
    if (tid < 32) {                      // wave 0; lanes 16-31 mirror 0-15
      const int m = tid & 15;
      float lw  = lout[m];
      float mx  = half16_max(lw);
      float lz  = mx + __logf(half16_sum(__expf(lw - mx)));   // log_softmax denom
      float comp = 0.f;
      #pragma unroll
      for (int d = 0; d < 4; ++d) {
        float mean = lout[16 + m * 4 + d];
        float lv   = lout[80 + m * 4 + d];
        float df   = y[d] - mean;
        comp = fmaf(-0.5f, LOG_2PI + lv + df * df * __expf(-lv), comp);
      }
      float tot = (lw - lz) + comp;
      float mx2 = half16_max(tot);
      float lse = mx2 + __logf(half16_sum(__expf(tot - mx2)));
      if (tid == 0) logp_sum += lse;
    }
    __syncthreads();
    // ---- jump update with same y_t ----
    apply_jump(p, y, z, x16, actA16, actB16, dh);
  }

  if (tid == 0) *out = logp_sum;
}

// ---------------------------------------------------------------------------
// Host launch: 12 weight-repack kernels, then the persistent scan kernel.
// d_in order: obs, F(W1,b1,W2,b2,W3,b3), G(...), W(...), L(...), c0, dt, ns
// ---------------------------------------------------------------------------
extern "C" void kernel_launch(void* const* d_in, const int* in_sizes, int n_in,
                              void* d_out, int out_size, void* d_ws, size_t ws_size,
                              hipStream_t stream) {
  (void)n_in; (void)out_size; (void)ws_size;
  Params p;
  p.obs = (const float*)d_in[0];
  p.T   = in_sizes[0] / 4;

  _Float16* ws = (_Float16*)d_ws;
  size_t off = 0;
  auto conv = [&](int idx, int outD, int inD, int outP, int inP) -> const _Float16* {
    _Float16* dst = ws + off;
    int total = outP * inP;
    off += (size_t)total;
    hipLaunchKernelGGL(convert_weights, dim3((total + 255) / 256), dim3(256), 0, stream,
                       (const float*)d_in[idx], dst, outD, inD, outP, inP);
    return dst;
  };

  p.WF1 = conv(1,  256, 256, 256, 256);  p.bF1 = (const float*)d_in[2];
  p.WF2 = conv(3,  256, 256, 256, 256);  p.bF2 = (const float*)d_in[4];
  p.WF3 = conv(5,  128, 256, 128, 256);  p.bF3 = (const float*)d_in[6];
  p.WG1 = conv(7,  256, 128, 256, 128);  p.bG1 = (const float*)d_in[8];
  p.WG2 = conv(9,  256, 256, 256, 256);  p.bG2 = (const float*)d_in[10];
  p.WG3 = conv(11, 128, 256, 128, 256);  p.bG3 = (const float*)d_in[12];
  p.WW1 = conv(13, 256, 132, 256, 160);  p.bW1 = (const float*)d_in[14];
  p.WW2 = conv(15, 256, 256, 256, 256);  p.bW2 = (const float*)d_in[16];
  p.WW3 = conv(17, 128, 256, 128, 256);  p.bW3 = (const float*)d_in[18];
  p.WL1 = conv(19, 256, 256, 256, 256);  p.bL1 = (const float*)d_in[20];
  p.WL2 = conv(21, 256, 256, 256, 256);  p.bL2 = (const float*)d_in[22];
  p.WL3 = conv(23, 144, 256, 160, 256);  p.bL3 = (const float*)d_in[24];
  p.c0  = (const float*)d_in[25];
  p.dtp = (const float*)d_in[26];
  p.nsp = (const int*)d_in[27];

  hipLaunchKernelGGL(njsde_main, dim3(1), dim3(512), 0, stream, p, (float*)d_out);
}
